// MimiEncoder_16896401342949
// MI455X (gfx1250) — compile-verified, matrix operand order checked by reading, not run on hardware
//
#include <hip/hip_runtime.h>
#include <hip/hip_bf16.h>

// ---------------------------------------------------------------------------
// Types for CDNA5 WMMA (gfx1250, wave32)
// ---------------------------------------------------------------------------
typedef __bf16 v16bf __attribute__((ext_vector_type(16)));
typedef float  v8f   __attribute__((ext_vector_type(8)));

#define BM 128   // block tile rows (time)
#define BN 64    // block tile cols (Cout)
#define KC 32    // bf16 K per WMMA
#define LDA 40   // padded LDS row length (bf16): 80B stride -> conflict-free, 16B aligned
#define LDB 40

// ---------------------------------------------------------------------------
// f32 -> bf16 conversion
// ---------------------------------------------------------------------------
__global__ __launch_bounds__(256) void f32_to_bf16_kernel(
    const float* __restrict__ src, __bf16* __restrict__ dst, int n) {
  int i = blockIdx.x * 256 + threadIdx.x;
  if (i < n) dst[i] = (__bf16)src[i];
}

// ---------------------------------------------------------------------------
// Generic causal-conv / GEMM via v_wmma_f32_16x16x32_bf16.
// Block = 128x64 output tile, 8 waves; each wave owns a 16x64 strip and issues
// 4 WMMAs per K-chunk from one A fragment + 4 B fragments, each fragment read
// as two ds_load_b128 from LDS. B is staged transposed (Bt[n][k]) so both
// fragment types are contiguous. kw==1 && stride==1 takes a fully vectorized
// uint4 staging path (all transformer GEMMs and 1x1 convs).
// Epilogue: (+bias[n]) * gamma[n] + resid_f32 + resid_bf16 -> out_bf16/out_f32
// ---------------------------------------------------------------------------
__global__ __launch_bounds__(256) void conv_wmma_kernel(
    const __bf16* __restrict__ X, const __bf16* __restrict__ W,
    const float* __restrict__ bias, const __bf16* __restrict__ resid_bf,
    const float* __restrict__ resid_f, const float* __restrict__ gamma,
    __bf16* __restrict__ out_bf, float* __restrict__ out_f,
    int T_in, int T_out, int Cin, int cin_sh, int Cout,
    int kw, int stride, int dilation, int elu_in) {
  __shared__ alignas(16) __bf16 As[BM][LDA];
  __shared__ alignas(16) __bf16 Bt[BN][LDB];

  const int b    = blockIdx.z;
  const int t0   = blockIdx.x * BM;
  const int n0   = blockIdx.y * BN;
  const int tid  = threadIdx.x;
  const int lane = tid & 31;
  const int wave = tid >> 5;          // 8 waves, wave w -> rows [w*16, w*16+16)
  const int hi   = lane >> 4;
  const int lr   = lane & 15;

  const __bf16* Xb = X + (size_t)b * T_in * Cin;
  const int Ktot = kw * Cin;
  const bool gemm_mode = (kw == 1) && (stride == 1);

  v8f acc[4];
#pragma unroll
  for (int c2 = 0; c2 < 4; ++c2)
#pragma unroll
    for (int r = 0; r < 8; ++r) acc[c2][r] = 0.f;

  for (int k0 = 0; k0 < Ktot; k0 += KC) {
    // ---- stage A tile (128 rows x 32 K) ----
    if (gemm_mode) {
      // vector path: thread -> (row m = tid>>1, half = tid&1), 16 bf16 per thread
      int m    = tid >> 1;
      int half = tid & 1;
      int to   = t0 + m;
      uint4 c0 = make_uint4(0u, 0u, 0u, 0u);
      uint4 c1 = make_uint4(0u, 0u, 0u, 0u);
      if (to < T_out) {
        const uint4* src = (const uint4*)(Xb + (size_t)to * Cin + k0 + half * 16);
        c0 = src[0];
        c1 = src[1];
      }
      if (elu_in) {
        union { uint4 u[2]; __bf16 h[16]; } t;
        t.u[0] = c0; t.u[1] = c1;
#pragma unroll
        for (int e = 0; e < 16; ++e) {
          float v = (float)t.h[e];
          t.h[e] = (__bf16)(v > 0.f ? v : (__expf(v) - 1.f));
        }
        c0 = t.u[0]; c1 = t.u[1];
      }
      uint4* dst = (uint4*)&As[m][half * 16];
      dst[0] = c0;
      dst[1] = c1;
    } else {
      // gathered path (strided / dilated / multi-tap causal conv)
      for (int i = tid; i < BM * KC; i += 256) {
        int m  = i >> 5;
        int kk = i & 31;
        int kg = k0 + kk;
        float v = 0.f;
        if (kg < Ktot) {
          int kwi = kg >> cin_sh;        // Cin is a power of two
          int ci2 = kg & (Cin - 1);
          int to  = t0 + m;
          int pos = to * stride - dilation * (kw - 1 - kwi);
          if (to < T_out && pos >= 0 && pos < T_in) {
            v = (float)Xb[(size_t)pos * Cin + ci2];
            if (elu_in) v = v > 0.f ? v : (__expf(v) - 1.f);
          }
        }
        As[m][kk] = (__bf16)v;
      }
    }

    // ---- stage B tile transposed: Bt[n][kk], one uint4 global load/thread ----
    {
      int kk = tid >> 3;          // 0..31
      int n8 = (tid & 7) * 8;     // 0..56  (Cout is always a multiple of 64)
      int kg = k0 + kk;
      union { uint4 u; __bf16 h[8]; } t;
      t.u = make_uint4(0u, 0u, 0u, 0u);
      if (kg < Ktot) t.u = *(const uint4*)(W + (size_t)kg * Cout + n0 + n8);
#pragma unroll
      for (int e = 0; e < 8; ++e) Bt[n8 + e][kk] = t.h[e];
    }
    __syncthreads();

    // prefetch next weight K-chunk into L2 (global_prefetch_b8)
    if (k0 + KC < Ktot)
      __builtin_prefetch(W + (size_t)(k0 + KC + (tid >> 3)) * Cout + n0 + (tid & 7) * 8, 0, 1);

    // ---- fragments: two ds_load_b128 each (ISA 7.12.2 16-bit A layout) ----
    const int mrow = wave * 16 + lr;
    v16bf av;
    {
      union { uint4 u[2]; v16bf v; } t;
      t.u[0] = *(const uint4*)&As[mrow][hi * 8];
      t.u[1] = *(const uint4*)&As[mrow][16 + hi * 8];
      av = t.v;
    }
#pragma unroll
    for (int c2 = 0; c2 < 4; ++c2) {
      v16bf bv;
      union { uint4 u[2]; v16bf v; } t;
      const int ncol = c2 * 16 + lr;
      t.u[0] = *(const uint4*)&Bt[ncol][hi * 8];
      t.u[1] = *(const uint4*)&Bt[ncol][16 + hi * 8];
      bv = t.v;
      acc[c2] = __builtin_amdgcn_wmma_f32_16x16x32_bf16(
          false, av, false, bv, (short)0, acc[c2], false, false);
    }
    __syncthreads();
  }

  // ---- epilogue: D element r at (M = hi*8 + r, N = lr) of each 16x16 tile ----
  const int mbase = t0 + wave * 16 + hi * 8;
#pragma unroll
  for (int c2 = 0; c2 < 4; ++c2) {
    const int colg = n0 + c2 * 16 + lr;
    if (colg < Cout) {
      const float bia = bias ? bias[colg] : 0.f;
      const float gv  = gamma ? gamma[colg] : 1.f;
#pragma unroll
      for (int r = 0; r < 8; ++r) {
        int rowg = mbase + r;
        if (rowg < T_out) {
          size_t idx = ((size_t)b * T_out + rowg) * Cout + colg;
          float v = (acc[c2][r] + bia) * gv;
          if (resid_f)  v += resid_f[idx];
          if (resid_bf) v += (float)resid_bf[idx];
          if (out_bf) out_bf[idx] = (__bf16)v;
          if (out_f)  out_f[idx]  = v;
        }
      }
    }
  }
}

// ---------------------------------------------------------------------------
// LayerNorm: one block per row, f32 in -> normalized bf16 out
// ---------------------------------------------------------------------------
__global__ __launch_bounds__(256) void ln_kernel(
    const float* __restrict__ X, const float* __restrict__ g,
    const float* __restrict__ bta, __bf16* __restrict__ out, int D) {
  __shared__ float red[8];
  __shared__ float stat[2];
  const int row = blockIdx.x;
  const float* x = X + (size_t)row * D;

  float s = 0.f;
  for (int i = threadIdx.x; i < D; i += 256) s += x[i];
#pragma unroll
  for (int off = 16; off > 0; off >>= 1) s += __shfl_down(s, off);
  if ((threadIdx.x & 31) == 0) red[threadIdx.x >> 5] = s;
  __syncthreads();
  if (threadIdx.x == 0) {
    float t = 0.f;
    for (int i = 0; i < 8; ++i) t += red[i];
    stat[0] = t / (float)D;
  }
  __syncthreads();
  const float mean = stat[0];

  float v = 0.f;
  for (int i = threadIdx.x; i < D; i += 256) {
    float d = x[i] - mean;
    v += d * d;
  }
#pragma unroll
  for (int off = 16; off > 0; off >>= 1) v += __shfl_down(v, off);
  if ((threadIdx.x & 31) == 0) red[threadIdx.x >> 5] = v;
  __syncthreads();
  if (threadIdx.x == 0) {
    float t = 0.f;
    for (int i = 0; i < 8; ++i) t += red[i];
    stat[1] = rsqrtf(t / (float)D + 1e-6f);
  }
  __syncthreads();
  const float rstd = stat[1];

  for (int i = threadIdx.x; i < D; i += 256)
    out[(size_t)row * D + i] = (__bf16)((x[i] - mean) * rstd * g[i] + bta[i]);
}

// ---------------------------------------------------------------------------
// Causal attention with fused RoPE; one block per (batch, head).
// qkv: [B*T, 3*Dm] bf16 ; out: [B*T, Dm] bf16. Online softmax, K/V in LDS.
// ---------------------------------------------------------------------------
#define ATT_DH 64
#define ATT_TMAX 250
__global__ __launch_bounds__(256) void attn_kernel(
    const __bf16* __restrict__ qkv, __bf16* __restrict__ out, int T, int Dm) {
  __shared__ __bf16 Ks[ATT_TMAX * ATT_DH];
  __shared__ __bf16 Vs[ATT_TMAX * ATT_DH];
  const int bh  = blockIdx.x;
  const int b   = bh >> 3;
  const int h   = bh & 7;
  const int tid = threadIdx.x;
  const size_t rowStride = (size_t)3 * Dm;
  const __bf16* base = qkv + (size_t)b * T * rowStride;

  // stage RoPE'd K and raw V
  for (int idx = tid; idx < T * ATT_DH; idx += 256) {
    int t = idx >> 6;
    int d = idx & 63;
    const __bf16* krow = base + (size_t)t * rowStride + Dm + h * ATT_DH;
    const __bf16* vrow = base + (size_t)t * rowStride + 2 * Dm + h * ATT_DH;
    float kx  = (float)krow[d];
    int   dp  = (d < 32) ? d + 32 : d - 32;
    float par = (float)krow[dp];
    float rot = (d < 32) ? -par : par;
    float fr  = __powf(10000.f, -(float)(d & 31) / 32.f);
    float sn, cs;
    __sincosf((float)t * fr, &sn, &cs);
    Ks[idx] = (__bf16)(kx * cs + rot * sn);
    Vs[idx] = vrow[d];
  }
  __syncthreads();

  const float scale = 0.125f;  // 64^-0.5
  for (int t = tid; t < T; t += 256) {
    const __bf16* qrow = base + (size_t)t * rowStride + h * ATT_DH;
    float q[ATT_DH];
#pragma unroll
    for (int d = 0; d < ATT_DH; ++d) {
      float qx  = (float)qrow[d];
      int   dp  = (d < 32) ? d + 32 : d - 32;
      float par = (float)qrow[dp];
      float rot = (d < 32) ? -par : par;
      float fr  = __powf(10000.f, -(float)(d & 31) / 32.f);
      float sn, cs;
      __sincosf((float)t * fr, &sn, &cs);
      q[d] = (qx * cs + rot * sn) * scale;
    }
    float mx = -3.0e38f, l = 0.f;
    float o[ATT_DH];
#pragma unroll
    for (int d = 0; d < ATT_DH; ++d) o[d] = 0.f;
    for (int j = 0; j <= t; ++j) {
      float s = 0.f;
#pragma unroll
      for (int d = 0; d < ATT_DH; ++d) s += q[d] * (float)Ks[j * ATT_DH + d];
      float mn   = fmaxf(mx, s);
      float corr = __expf(mx - mn);
      float p    = __expf(s - mn);
      l = l * corr + p;
#pragma unroll
      for (int d = 0; d < ATT_DH; ++d)
        o[d] = o[d] * corr + p * (float)Vs[j * ATT_DH + d];
      mx = mn;
    }
    float inv = 1.f / l;
    __bf16* orow = out + ((size_t)(b * T + t)) * Dm + h * ATT_DH;
#pragma unroll
    for (int d = 0; d < ATT_DH; ++d) orow[d] = (__bf16)(o[d] * inv);
  }
}

// ---------------------------------------------------------------------------
// gelu(g) * u -> bf16 (tanh-approx gelu, matching jax.nn.gelu default)
// ---------------------------------------------------------------------------
__global__ __launch_bounds__(256) void gelu_mul_kernel(
    const float* __restrict__ g, const float* __restrict__ u,
    __bf16* __restrict__ out, int n) {
  int i = blockIdx.x * 256 + threadIdx.x;
  if (i < n) {
    float x  = g[i];
    float t  = 0.7978845608028654f * (x + 0.044715f * x * x * x);
    float ge = 0.5f * x * (1.f + tanhf(t));
    out[i] = (__bf16)(ge * u[i]);
  }
}

// ---------------------------------------------------------------------------
// Host orchestration
// ---------------------------------------------------------------------------
extern "C" void kernel_launch(void* const* d_in, const int* in_sizes, int n_in,
                              void* d_out, int out_size, void* d_ws, size_t ws_size,
                              hipStream_t stream) {
  (void)in_sizes; (void)n_in; (void)out_size; (void)ws_size;

  // ---- input enumeration (setup_inputs insertion order) ----
  int ci = 0;
  auto nextp = [&]() -> const float* { return (const float*)d_in[ci++]; };

  const float* x_f = nextp();
  const float* icw = nextp();
  const float* icb = nextp();
  struct RB { const float *w1, *b1, *w2, *b2; };
  struct BLK { RB rb[2]; const float *dw, *db; };
  BLK blk[4];
  const int chs[5]     = {64, 128, 256, 512, 512};
  const int strides[4] = {4, 5, 6, 8};
  for (int i = 0; i < 4; ++i) {
    for (int j = 0; j < 2; ++j) {
      blk[i].rb[j].w1 = nextp(); blk[i].rb[j].b1 = nextp();
      blk[i].rb[j].w2 = nextp(); blk[i].rb[j].b2 = nextp();
    }
    blk[i].dw = nextp(); blk[i].db = nextp();
  }
  const float* pqw = nextp();
  struct LYR { const float *ln1g, *ln1b, *wqkv, *wo, *ga, *ln2g, *ln2b, *wg, *wu, *wd, *gm; };
  LYR ly[8];
  for (int l = 0; l < 8; ++l) {
    ly[l].ln1g = nextp(); ly[l].ln1b = nextp();
    ly[l].wqkv = nextp(); ly[l].wo   = nextp(); ly[l].ga = nextp();
    ly[l].ln2g = nextp(); ly[l].ln2b = nextp();
    ly[l].wg   = nextp(); ly[l].wu   = nextp(); ly[l].wd = nextp();
    ly[l].gm   = nextp();
  }
  const float* ng  = nextp();
  const float* nb  = nextp();
  const float* opw = nextp();

  // ---- workspace carving ----
  char*  wp  = (char*)d_ws;
  size_t off = 0;
  auto alloc = [&](size_t bytes) -> void* {
    void* p = wp + off;
    off += (bytes + 255) & ~(size_t)255;
    return p;
  };

  const size_t bigElems = (size_t)2 * 240000 * 64;
  __bf16* bufA = (__bf16*)alloc(bigElems * sizeof(__bf16));
  __bf16* bufB = (__bf16*)alloc(bigElems * sizeof(__bf16));
  __bf16* bufC = (__bf16*)alloc(bigElems * sizeof(__bf16));

  // ring of bf16 weight-conversion slots (stream order serializes reuse)
  const size_t maxW = 4194304;  // largest weight: 16*512*512
  __bf16* wring[4];
  for (int k = 0; k < 4; ++k) wring[k] = (__bf16*)alloc(maxW * sizeof(__bf16));
  int wslot = 0;
  auto tobf = [&](const float* src, size_t n) -> __bf16* {
    __bf16* dst = wring[wslot];
    wslot = (wslot + 1) & 3;
    f32_to_bf16_kernel<<<dim3((unsigned)((n + 255) / 256)), dim3(256), 0, stream>>>(
        src, dst, (int)n);
    return dst;
  };

  auto ilog2 = [](int v) -> int { int s = 0; while ((1 << s) < v) ++s; return s; };

  auto conv = [&](const __bf16* X, const __bf16* W, const float* bias,
                  const __bf16* resid_bf, const float* resid_f, const float* gamma,
                  __bf16* out_bf, float* out_f,
                  int Bn, int T_in, int T_out, int Cin, int Cout,
                  int kw, int stride, int dil, int elu) {
    dim3 grid((T_out + BM - 1) / BM, (Cout + BN - 1) / BN, Bn);
    conv_wmma_kernel<<<grid, 256, 0, stream>>>(
        X, W, bias, resid_bf, resid_f, gamma, out_bf, out_f,
        T_in, T_out, Cin, ilog2(Cin), Cout, kw, stride, dil, elu);
  };

  // ---- conv stack ----
  __bf16* xbf = tobf(x_f, 480000);  // [2, 240000, 1]
  {
    __bf16* w = tobf(icw, (size_t)7 * 1 * 64);
    conv(xbf, w, icb, nullptr, nullptr, nullptr, bufA, nullptr,
         2, 240000, 240000, 1, 64, 7, 1, 1, 0);
  }

  __bf16 *h = bufA, *u = bufB, *h2 = bufC;
  int T = 240000, C = 64;
  for (int i = 0; i < 4; ++i) {
    for (int j = 0; j < 2; ++j) {
      int dl = (j == 0) ? 1 : 2;
      __bf16* w1 = tobf(blk[i].rb[j].w1, (size_t)3 * C * C);
      conv(h, w1, blk[i].rb[j].b1, nullptr, nullptr, nullptr, u, nullptr,
           2, T, T, C, C, 3, 1, dl, 1);
      __bf16* w2 = tobf(blk[i].rb[j].w2, (size_t)C * C);
      conv(u, w2, blk[i].rb[j].b2, h, nullptr, nullptr, h2, nullptr,
           2, T, T, C, C, 1, 1, 1, 1);
      __bf16* tmp = h; h = h2; h2 = tmp;
    }
    int s  = strides[i];
    int Co = chs[(i + 1 > 4) ? 4 : i + 1];
    int Td = T / s;
    __bf16* wd = tobf(blk[i].dw, (size_t)2 * s * C * Co);
    conv(h, wd, blk[i].db, nullptr, nullptr, nullptr, u, nullptr,
         2, T, Td, C, Co, 2 * s, s, 1, 1);
    __bf16* tmp = h; h = u; u = tmp;
    T = Td; C = Co;
  }
  // now h: bf16 [2, 250, 512]

  // ---- transformer scratch ----
  const int ROWS = 2 * 250;  // 500
  float*  hres = (float*)alloc((size_t)ROWS * 512 * sizeof(float));
  __bf16* xn   = (__bf16*)alloc((size_t)ROWS * 512 * sizeof(__bf16));
  __bf16* qkvb = (__bf16*)alloc((size_t)ROWS * 1536 * sizeof(__bf16));
  __bf16* attb = (__bf16*)alloc((size_t)ROWS * 512 * sizeof(__bf16));
  float*  gf   = (float*)alloc((size_t)ROWS * 2048 * sizeof(float));
  float*  uf   = (float*)alloc((size_t)ROWS * 2048 * sizeof(float));
  __bf16* mb   = (__bf16*)alloc((size_t)ROWS * 2048 * sizeof(__bf16));

  // pre-quant: h_f32 = elu(h) @ pre_quant_proj  (no bias)
  {
    __bf16* pq = tobf(pqw, (size_t)512 * 512);
    conv(h, pq, nullptr, nullptr, nullptr, nullptr, nullptr, hres,
         1, ROWS, ROWS, 512, 512, 1, 1, 1, 1);
  }

  // ---- transformer layers ----
  for (int l = 0; l < 8; ++l) {
    const LYR& p = ly[l];
    ln_kernel<<<ROWS, 256, 0, stream>>>(hres, p.ln1g, p.ln1b, xn, 512);
    __bf16* wqkv = tobf(p.wqkv, (size_t)512 * 1536);
    conv(xn, wqkv, nullptr, nullptr, nullptr, nullptr, qkvb, nullptr,
         1, ROWS, ROWS, 512, 1536, 1, 1, 1, 0);
    attn_kernel<<<16, 256, 0, stream>>>(qkvb, attb, 250, 512);
    __bf16* wo = tobf(p.wo, (size_t)512 * 512);
    conv(attb, wo, nullptr, nullptr, hres, p.ga, nullptr, hres,
         1, ROWS, ROWS, 512, 512, 1, 1, 1, 0);  // h += (attn @ wo) * gamma_attn
    ln_kernel<<<ROWS, 256, 0, stream>>>(hres, p.ln2g, p.ln2b, xn, 512);
    __bf16* wg = tobf(p.wg, (size_t)512 * 2048);
    conv(xn, wg, nullptr, nullptr, nullptr, nullptr, nullptr, gf,
         1, ROWS, ROWS, 512, 2048, 1, 1, 1, 0);
    __bf16* wu = tobf(p.wu, (size_t)512 * 2048);
    conv(xn, wu, nullptr, nullptr, nullptr, nullptr, nullptr, uf,
         1, ROWS, ROWS, 512, 2048, 1, 1, 1, 0);
    gelu_mul_kernel<<<dim3((ROWS * 2048 + 255) / 256), dim3(256), 0, stream>>>(
        gf, uf, mb, ROWS * 2048);
    __bf16* wd = tobf(p.wd, (size_t)2048 * 512);
    conv(mb, wd, nullptr, nullptr, hres, p.gm, nullptr, hres,
         1, ROWS, ROWS, 2048, 512, 1, 1, 1, 0);  // h += mlp * gamma_mlp
  }

  // ---- final LN + output projection ----
  ln_kernel<<<ROWS, 256, 0, stream>>>(hres, ng, nb, xn, 512);
  {
    __bf16* wo = tobf(opw, (size_t)512 * 256);
    conv(xn, wo, nullptr, nullptr, nullptr, nullptr, nullptr, (float*)d_out,
         1, ROWS, ROWS, 512, 256, 1, 1, 1, 0);
  }
}
